// EncoderAttention_7138235646158
// MI455X (gfx1250) — matrix operand order repaired and checked by reading, not compiled, not run on hardware
//
#include <hip/hip_runtime.h>
#include <hip/hip_bf16.h>

typedef __attribute__((ext_vector_type(16))) _Float16 v16h;
typedef __attribute__((ext_vector_type(8)))  float    v8f;

#define B_SZ   8
#define L_SEQ  1025
#define HID    1024
#define NH     16
#define NKV    4
#define DH     64
#define L_PAD  1056            // 33 * 32
#define MROWS  (B_SZ * L_SEQ)  // 8200
#define MP     8256            // 129 * 64 (padded GEMM rows)
#define NQKV   1536
#define KTILES 33              // key tiles of 32 -> 1056 keys
#define QTILES 65              // ceil(1025/16)

#define WMMA_F16(A_, B_, C_) \
  __builtin_amdgcn_wmma_f32_16x16x32_f16(false, (A_), false, (B_), (short)0, (C_), false, false)

// ---- async global->LDS copy (16B per lane) via inline asm (VFLAT async encoding).
// vdst = per-lane 32-bit LDS byte address; generic LDS pointers carry the LDS
// offset in their low 32 bits (ISA 10.2 aperture rules), so truncation is exact.
static __device__ __forceinline__ void cp_async16(const _Float16* g, _Float16* l) {
  unsigned lds_addr = (unsigned)(size_t)l;
  asm volatile("global_load_async_to_lds_b128 %0, %1, off"
               :: "v"(lds_addr), "v"(g)
               : "memory");
}
static __device__ __forceinline__ void wait_async_all() {
  asm volatile("s_wait_asynccnt 0x0" ::: "memory");
}

// ---- A fragment (16x32 f16, M x K), contiguous k order used here (k = 16h + s), not AMD's table order:
// lane 0-15: M=lane, K in {0..15}; lane 16-31: M=lane-16, K in {16..31}
// element pairs are K-consecutive -> dword (pair) loads per lane.
static __device__ __forceinline__ v16h load_afrag(const _Float16* __restrict__ src,
                                                  int ld, int row0, int k0) {
  int lane = threadIdx.x & 31;
  int m = row0 + (lane & 15);
  int koff = (lane >> 4) << 4;  // +16 for upper half-wave
  const unsigned* p = reinterpret_cast<const unsigned*>(src + (size_t)m * ld + k0 + koff);
  union { v16h v; unsigned u[8]; } a;
#pragma unroll
  for (int j = 0; j < 8; ++j) a.u[j] = p[j];
  return a.v;
}

// ---- B fragment (32x16 f16, K x N) where memory holds B^T row-major: src[n*ld + k].
// lane 0-15: N=lane, K=k0..k0+15; lane 16-31: N=lane-16, K=k0+16..k0+31 -> two b128 loads.
static __device__ __forceinline__ v16h load_bfragT(const _Float16* __restrict__ src,
                                                   int ld, int col0, int k0) {
  int lane = threadIdx.x & 31;
  int n = col0 + (lane & 15);
  int koff = (lane >> 4) << 4;  // +16 for upper half-wave
  const uint4* p = reinterpret_cast<const uint4*>(src + (size_t)n * ld + k0 + koff);
  union { v16h v; uint4 q[2]; } b;
  b.q[0] = p[0];
  b.q[1] = p[1];
  return b.v;
}

// ---------------- convert / pad kernels ----------------
__global__ void ea_cvt_x(const float* __restrict__ X, _Float16* __restrict__ Xh,
                         _Float16* __restrict__ Ah) {
  size_t i = (size_t)blockIdx.x * blockDim.x + threadIdx.x;
  if (i >= (size_t)MP * HID) return;
  int row = (int)(i >> 10);
  if (row < MROWS) {
    Xh[i] = (_Float16)X[i];
  } else {
    Xh[i] = (_Float16)0.f;
    Ah[i] = (_Float16)0.f;  // zero pad rows of attention staging buffer
  }
}

__global__ void ea_cvt_w(const float* __restrict__ Wq, const float* __restrict__ Wk,
                         const float* __restrict__ Wv, const float* __restrict__ Wo,
                         _Float16* __restrict__ Wqkv, _Float16* __restrict__ Woh) {
  size_t i = (size_t)blockIdx.x * blockDim.x + threadIdx.x;
  const size_t NQK = (size_t)NQKV * HID;
  const size_t NO  = (size_t)HID * (NH * DH);
  if (i < NQK) {
    int row = (int)(i >> 10), col = (int)(i & 1023);
    float v;
    if (row < 1024)      v = Wq[i];
    else if (row < 1280) v = Wk[(size_t)(row - 1024) * HID + col];
    else                 v = Wv[(size_t)(row - 1280) * HID + col];
    Wqkv[i] = (_Float16)v;
  } else if (i < NQK + NO) {
    size_t j = i - NQK;
    Woh[j] = (_Float16)Wo[j];
  }
}

// ---------------- generic f16 GEMM: C[m,n] = sum_k A[m,k] * Bw[n,k] ----------------
// Block = 4 waves. Double-buffered async-LDS pipeline: the async engine fetches
// tile k+1 (A 64x32 + B 64x32) while WMMAs consume tile k from LDS. Each tile is
// fetched once per block (no cross-wave duplication).
__global__ __launch_bounds__(128) void ea_gemm(const _Float16* __restrict__ A,
                                               const _Float16* __restrict__ Bw,
                                               float* __restrict__ C,
                                               int N, int K, int Mvalid) {
  __shared__ _Float16 atile[2][64 * 32];  // 4 KB each
  __shared__ _Float16 btile[2][64 * 32];  // 4 KB each
  int tid  = (int)threadIdx.x;
  int wave = tid >> 5;
  int lane = tid & 31;
  int row0b = blockIdx.x * 64;
  int col0  = blockIdx.y * 64;
  // staging map: thread t covers row t/2, halves [(t&1)*16, +16) => 2 x b128
  int srow = tid >> 1;
  int scol = (tid & 1) << 4;
  const _Float16* ag = A  + (size_t)(row0b + srow) * K + scol;
  const _Float16* bg = Bw + (size_t)(col0  + srow) * K + scol;
  int sl = srow * 32 + scol;

  auto stage = [&](int buf, int kk) {
    cp_async16(ag + kk,     &atile[buf][sl]);
    cp_async16(ag + kk + 8, &atile[buf][sl + 8]);
    cp_async16(bg + kk,     &btile[buf][sl]);
    cp_async16(bg + kk + 8, &btile[buf][sl + 8]);
  };

  v8f c0 = {}, c1 = {}, c2 = {}, c3 = {};
  stage(0, 0);
  wait_async_all();
  __syncthreads();
  int cur = 0;
  for (int kk = 0; kk < K; kk += 32) {
    if (kk + 32 < K) stage(cur ^ 1, kk + 32);  // overlap with compute below
    v16h a = load_afrag(&atile[cur][0], 32, wave * 16, 0);
    c0 = WMMA_F16(a, load_bfragT(&btile[cur][0], 32, 0,  0), c0);
    c1 = WMMA_F16(a, load_bfragT(&btile[cur][0], 32, 16, 0), c1);
    c2 = WMMA_F16(a, load_bfragT(&btile[cur][0], 32, 32, 0), c2);
    c3 = WMMA_F16(a, load_bfragT(&btile[cur][0], 32, 48, 0), c3);
    asm volatile("s_wait_dscnt 0" ::: "memory");  // reads/stores done before swap
    wait_async_all();
    __syncthreads();
    cur ^= 1;
  }
  int n = lane & 15, mb = (lane >> 4) << 3;
  int row0 = row0b + wave * 16;
#pragma unroll
  for (int r = 0; r < 8; ++r) {
    int row = row0 + mb + r;
    if (row < Mvalid) {
      float* cp = C + (size_t)row * N + col0 + n;
      cp[0]  = c0[r];
      cp[16] = c1[r];
      cp[32] = c2[r];
      cp[48] = c3[r];
    }
  }
}

// ---------------- RMS-norm + RoPE + layout change ----------------
// one wave per (b, l, head); lane d owns elements d and d+32 (rotate_half pairing).
// heads 0..15 -> Q, 16..19 -> K, 20..23 -> V (no norm/rope, stored transposed DHxL).
__global__ __launch_bounds__(32) void ea_normrope(const float* __restrict__ Y,
                                                  const float* __restrict__ qw,
                                                  const float* __restrict__ kw,
                                                  _Float16* __restrict__ Qh,
                                                  _Float16* __restrict__ Kh,
                                                  _Float16* __restrict__ Vt) {
  int bl = blockIdx.x;
  int b = bl / L_PAD, l = bl - b * L_PAD;
  int head = blockIdx.y;
  int lane = threadIdx.x;
  bool valid = (l < L_SEQ);
  int cb = (head < NH) ? head * DH
           : (head < NH + NKV) ? (NH * DH + (head - NH) * DH)
                               : (NH * DH + NKV * DH + (head - NH - NKV) * DH);
  float x1 = 0.f, x2 = 0.f;
  if (valid) {
    const float* yr = Y + (size_t)(b * L_SEQ + l) * NQKV + cb;
    x1 = yr[lane];
    x2 = yr[lane + 32];
  }
  if (head >= NH + NKV) {  // V: transpose-store, no norm
    int vh = head - NH - NKV;
    _Float16* vp = Vt + (size_t)(b * NKV + vh) * DH * L_PAD + l;
    vp[(size_t)lane * L_PAD] = (_Float16)x1;
    vp[(size_t)(lane + 32) * L_PAD] = (_Float16)x2;
    return;
  }
  float ss = x1 * x1 + x2 * x2;
#pragma unroll
  for (int mk = 16; mk >= 1; mk >>= 1) ss += __shfl_xor(ss, mk, 32);
  float inv = rsqrtf(ss * (1.0f / 64.0f) + 1e-6f);
  const float* w = (head < NH) ? qw : kw;
  x1 *= inv * w[lane];
  x2 *= inv * w[lane + 32];
  if (valid && l >= 1) {  // RoPE, pos = l-1; inv_freq = 10000^(-lane/32)
    float theta = (float)(l - 1) * __expf((float)lane * (-9.210340371976184f / 32.0f));
    float c = cosf(theta), s = sinf(theta);
    float o1 = x1 * c - x2 * s;
    float o2 = x2 * c + x1 * s;
    x1 = o1; x2 = o2;
  }
  if (!valid) { x1 = 0.f; x2 = 0.f; }
  _Float16* op = (head < NH)
      ? Qh + ((size_t)(b * NH + head) * L_PAD + l) * DH
      : Kh + ((size_t)(b * NKV + (head - NH)) * L_PAD + l) * DH;
  op[lane] = (_Float16)x1;
  op[lane + 32] = (_Float16)x2;
}

// ---------------- flash attention: one wave per 16-row query tile ----------------
__global__ __launch_bounds__(128) void ea_attn(const _Float16* __restrict__ Qh,
                                               const _Float16* __restrict__ Kh,
                                               const _Float16* __restrict__ Vt,
                                               _Float16* __restrict__ Ah) {
  __shared__ _Float16 pbuf[4][16 * 32];
  int wave = threadIdx.x >> 5;
  int lane = threadIdx.x & 31;
  int bh = blockIdx.x;                 // 0..127
  int b = bh >> 4, h = bh & 15;
  int kvh = h >> 2;                    // GQA: 4 Q heads per KV head
  int qt_raw = blockIdx.y * 4 + wave;
  bool wvalid = (qt_raw < QTILES);
  int qt = wvalid ? qt_raw : 0;
  const _Float16* Q  = Qh + (size_t)(b * NH + h) * L_PAD * DH;
  const _Float16* Kp = Kh + (size_t)(b * NKV + kvh) * L_PAD * DH;
  const _Float16* Vp = Vt + (size_t)(b * NKV + kvh) * DH * L_PAD;

  v16h qa0 = load_afrag(Q, DH, qt * 16, 0);
  v16h qa1 = load_afrag(Q, DH, qt * 16, 32);
  v8f o0 = {}, o1 = {}, o2 = {}, o3 = {};
  float mrow[8], lrow[8], al[8];
#pragma unroll
  for (int r = 0; r < 8; ++r) { mrow[r] = -3.0e38f; lrow[r] = 0.f; }
  _Float16* pb = &pbuf[wave][0];
  int coln = lane & 15, mb = (lane >> 4) << 3;

  for (int kt = 0; kt < KTILES; ++kt) {
    int kb = kt * 32;
    // S tile (16 x 32): scores = Q . K^T  (no 1/sqrt(d) in reference)
    v8f s0 = {}, s1 = {};
    s0 = WMMA_F16(qa0, load_bfragT(Kp, DH, kb,      0),  s0);
    s0 = WMMA_F16(qa1, load_bfragT(Kp, DH, kb,      32), s0);
    s1 = WMMA_F16(qa0, load_bfragT(Kp, DH, kb + 16, 0),  s1);
    s1 = WMMA_F16(qa1, load_bfragT(Kp, DH, kb + 16, 32), s1);
    bool bad0 = (kb + coln) >= L_SEQ;
    bool bad1 = (kb + 16 + coln) >= L_SEQ;
    // online softmax; C-layout row M = mb + r lives in this lane's 16-lane half,
    // so xor-masks 1..8 reduce exactly over the 16 columns of that row.
#pragma unroll
    for (int r = 0; r < 8; ++r) {
      if (bad0) s0[r] = -3.0e38f;
      if (bad1) s1[r] = -3.0e38f;
      float mx = fmaxf(s0[r], s1[r]);
#pragma unroll
      for (int mk = 8; mk >= 1; mk >>= 1) mx = fmaxf(mx, __shfl_xor(mx, mk, 32));
      float mn = fmaxf(mrow[r], mx);
      float a = __expf(mrow[r] - mn);
      mrow[r] = mn;
      float p0 = __expf(s0[r] - mn);
      float p1 = __expf(s1[r] - mn);
      float ps = p0 + p1;
#pragma unroll
      for (int mk = 8; mk >= 1; mk >>= 1) ps += __shfl_xor(ps, mk, 32);
      lrow[r] = lrow[r] * a + ps;
      al[r] = a;
      pb[(mb + r) * 32 + coln] = (_Float16)p0;        // stage P into LDS (f16)
      pb[(mb + r) * 32 + 16 + coln] = (_Float16)p1;
    }
#pragma unroll
    for (int r = 0; r < 8; ++r) {
      o0[r] *= al[r]; o1[r] *= al[r]; o2[r] *= al[r]; o3[r] *= al[r];
    }
    asm volatile("s_wait_dscnt 0" ::: "memory");      // cross-lane LDS write->read
    v16h pf = load_afrag(pb, 32, 0, 0);               // P as A fragment
    o0 = WMMA_F16(pf, load_bfragT(Vp, L_PAD, 0,  kb), o0);
    o1 = WMMA_F16(pf, load_bfragT(Vp, L_PAD, 16, kb), o1);
    o2 = WMMA_F16(pf, load_bfragT(Vp, L_PAD, 32, kb), o2);
    o3 = WMMA_F16(pf, load_bfragT(Vp, L_PAD, 48, kb), o3);
    asm volatile("s_wait_dscnt 0" ::: "memory");
  }
  if (!wvalid) return;
#pragma unroll
  for (int r = 0; r < 8; ++r) {
    int l = qt * 16 + mb + r;
    if (l < L_SEQ) {
      float inv = 1.0f / lrow[r];
      _Float16* op = Ah + (size_t)(b * L_SEQ + l) * (NH * DH) + h * DH;
      op[coln]      = (_Float16)(o0[r] * inv);
      op[16 + coln] = (_Float16)(o1[r] * inv);
      op[32 + coln] = (_Float16)(o2[r] * inv);
      op[48 + coln] = (_Float16)(o3[r] * inv);
    }
  }
}

extern "C" void kernel_launch(void* const* d_in, const int* in_sizes, int n_in,
                              void* d_out, int out_size, void* d_ws, size_t ws_size,
                              hipStream_t stream) {
  const float* X  = (const float*)d_in[0];
  const float* Wq = (const float*)d_in[1];
  const float* Wk = (const float*)d_in[2];
  const float* Wv = (const float*)d_in[3];
  const float* Wo = (const float*)d_in[4];
  const float* qw = (const float*)d_in[5];
  const float* kw = (const float*)d_in[6];
  float* out = (float*)d_out;

  // workspace carve-up (256B aligned chunks)
  char* w = (char*)d_ws;
  size_t off = 0;
  auto take = [&](size_t bytes) { char* p = w + off; off += (bytes + 255) & ~(size_t)255; return p; };
  _Float16* Xh   = (_Float16*)take((size_t)MP * HID * 2);
  _Float16* Wqkv = (_Float16*)take((size_t)NQKV * HID * 2);
  _Float16* Woh  = (_Float16*)take((size_t)HID * NH * DH * 2);
  float*    Yq   = (float*)   take((size_t)MP * NQKV * 4);
  _Float16* Qh   = (_Float16*)take((size_t)B_SZ * NH * L_PAD * DH * 2);
  _Float16* Kh   = (_Float16*)take((size_t)B_SZ * NKV * L_PAD * DH * 2);
  _Float16* Vt   = (_Float16*)take((size_t)B_SZ * NKV * DH * L_PAD * 2);
  _Float16* Ah   = (_Float16*)take((size_t)MP * NH * DH * 2);

  // 1) convert activations to f16 (+ zero pad rows of Ah)
  ea_cvt_x<<<(unsigned)(((size_t)MP * HID + 255) / 256), 256, 0, stream>>>(X, Xh, Ah);
  // 2) convert weights to f16 (Wq/Wk/Wv fused)
  {
    size_t tot = (size_t)NQKV * HID + (size_t)HID * NH * DH;
    ea_cvt_w<<<(unsigned)((tot + 255) / 256), 256, 0, stream>>>(Wq, Wk, Wv, Wo, Wqkv, Woh);
  }
  // 3) QKV projection: Yq[m, 0:1536] = Xh @ Wqkv^T
  ea_gemm<<<dim3(MP / 64, NQKV / 64), 128, 0, stream>>>(Xh, Wqkv, Yq, NQKV, HID, MP);
  // 4) RMS-norm + RoPE + head layout (Q/K) and V transpose
  ea_normrope<<<dim3(B_SZ * L_PAD, NH + 2 * NKV), 32, 0, stream>>>(Yq, qw, kw, Qh, Kh, Vt);
  // 5) flash attention -> Ah (f16, [B*L, H*DH])
  ea_attn<<<dim3(B_SZ * NH, (QTILES + 3) / 4), 128, 0, stream>>>(Qh, Kh, Vt, Ah);
  // 6) output projection: out = Ah @ Wo^T  (fp32, guarded to 8200 rows)
  ea_gemm<<<dim3(MP / 64, (NH * DH) / 64), 128, 0, stream>>>(Ah, Woh, out, HID, HID, MROWS);
}
